// MultiHeadAttention_18382460026906
// MI455X (gfx1250) — matrix operand order repaired and checked by reading, not compiled, not run on hardware
//
#include <hip/hip_runtime.h>
#include <hip/hip_bf16.h>

typedef __attribute__((ext_vector_type(16))) _Float16 v16h;
typedef __attribute__((ext_vector_type(8)))  _Float16 v8h;
typedef __attribute__((ext_vector_type(8)))  float    v8f;

#define DM   1024
#define NH   16
#define DK   64
#define BB   4
#define SS   2048
#define MROWS (BB*SS)          // 8192 total rows (b,s)

// ---------------- WMMA fragment helpers (CDNA5 16x16x32 f16 layouts) -----------

// A-matrix 16x32 f16: lane&15 = row M; half = lane>>4.
//   elements j=0..7  -> K = half*8 + j ; j=8..15 -> K = 16 + half*8 + (j-8)
static __device__ __forceinline__ v16h frag_a(const _Float16* row, int k0, int hh) {
  v8h lo = *(const v8h*)(row + k0 + hh * 8);
  v8h hi = *(const v8h*)(row + k0 + 16 + hh * 8);
  return __builtin_shufflevector(lo, hi, 0,1,2,3,4,5,6,7,8,9,10,11,12,13,14,15);
}

// B-matrix 32x16 f16 (row = column N of B, k-contiguous): j=0..7 -> K = half*8 + j ; j=8..15 -> K = 16 + half*8 + (j-8)
static __device__ __forceinline__ v16h frag_b(const _Float16* row, int k0, int hh) {
  v8h lo = *(const v8h*)(row + k0 + hh * 8);
  v8h hi = *(const v8h*)(row + k0 + 16 + hh * 8);
  return __builtin_shufflevector(lo, hi, 0,1,2,3,4,5,6,7,8,9,10,11,12,13,14,15);
}

static __device__ __forceinline__ v8f wmma_f16(v16h a, v16h b, v8f c) {
  return __builtin_amdgcn_wmma_f32_16x16x32_f16(false, a, false, b, (short)0, c, false, false);
}

// ---------------- async global->LDS staging (ASYNCcnt path) -------------------
// Generic LDS pointers: low 32 bits == LDS byte offset (ISA 10.2 truncation).
static __device__ __forceinline__ void async_ld_b128(const void* gaddr, void* laddr) {
  asm volatile("global_load_async_to_lds_b128 %0, %1, off"
               :: "v"((unsigned)(uintptr_t)laddr), "v"(gaddr) : "memory");
}
static __device__ __forceinline__ void wait_async0() {
  asm volatile("s_wait_asynccnt 0x0" ::: "memory");
}

// ---------------- DPP row reductions (16-lane rows == our reduce groups) -------

template <int CTRL>
static __device__ __forceinline__ float dpp_mov(float x) {
  int i = __builtin_amdgcn_update_dpp(0, __float_as_int(x), CTRL, 0xF, 0xF, true);
  return __int_as_float(i);
}
static __device__ __forceinline__ float rowred_max(float x) {
  x = fmaxf(x, dpp_mov<0x128>(x));   // ROW_ROR:8
  x = fmaxf(x, dpp_mov<0x124>(x));   // ROW_ROR:4
  x = fmaxf(x, dpp_mov<0x122>(x));   // ROW_ROR:2
  x = fmaxf(x, dpp_mov<0x121>(x));   // ROW_ROR:1
  return x;
}
static __device__ __forceinline__ float rowred_sum(float x) {
  x += dpp_mov<0x128>(x);
  x += dpp_mov<0x124>(x);
  x += dpp_mov<0x122>(x);
  x += dpp_mov<0x121>(x);
  return x;
}

// ---------------- conversion kernels ------------------------------------------

__global__ __launch_bounds__(256) void cvt_f16_kernel(const float* __restrict__ src,
                                                      _Float16* __restrict__ dst, int n) {
  int i = (blockIdx.x * blockDim.x + threadIdx.x) * 4;
  if (i < n) {
    float4 f = *(const float4*)(src + i);
    dst[i + 0] = (_Float16)f.x;
    dst[i + 1] = (_Float16)f.y;
    dst[i + 2] = (_Float16)f.z;
    dst[i + 3] = (_Float16)f.w;
  }
}

// Wt[n][k] = (f16) W[k][n]   (DM x DM)
__global__ __launch_bounds__(256) void cvtT_f16_kernel(const float* __restrict__ W,
                                                       _Float16* __restrict__ Wt) {
  int idx = blockIdx.x * blockDim.x + threadIdx.x;   // over DM*DM
  int k = idx & (DM - 1);
  int n = idx >> 10;
  Wt[(size_t)n * DM + k] = (_Float16)W[(size_t)k * DM + n];
}

// ---------------- projection / output GEMM ------------------------------------
// C[M=8192,N=1024] = A[M,K] (f16 rowmajor) @ Bt[N,K] (f16 n-major) + bias
// Block tile 128x128, BK=32, DOUBLE-BUFFERED async global->LDS staging.
// 8 waves, each 32x64 (2x4 wmma tiles), 8 wmma per stage, 32 stages.
template <int MODE>
__global__ __launch_bounds__(256) void gemm_kernel(const _Float16* __restrict__ A,
                                                   const _Float16* __restrict__ Bt,
                                                   const float* __restrict__ bias,
                                                   void* __restrict__ outp, float scale) {
  __shared__ _Float16 As[2][128][40];   // 2 x 128x32, stride 80B (16B aligned)
  __shared__ _Float16 Bs[2][128][40];

  const int m0   = blockIdx.x * 128;
  const int n0   = blockIdx.y * 128;
  const int tid  = threadIdx.x;
  const int lane = tid & 31;
  const int wave = tid >> 5;          // 0..7
  const int wm   = wave & 3;          // 4 m-groups of 32 rows
  const int wn   = wave >> 2;         // 2 n-groups of 64 cols
  const int hh   = lane >> 4;
  const int l15  = lane & 15;

  const int sr = tid >> 1;            // staging row 0..127
  const int so = (tid & 1) * 16;      // staging col offset (halfs)

  v8f acc[2][4] = {};

  // prologue: stage tile 0 into buffer 0
  {
    const _Float16* ga = A  + (size_t)(m0 + sr) * DM + so;
    const _Float16* gb = Bt + (size_t)(n0 + sr) * DM + so;
    async_ld_b128(ga,     &As[0][sr][so]);
    async_ld_b128(ga + 8, &As[0][sr][so + 8]);
    async_ld_b128(gb,     &Bs[0][sr][so]);
    async_ld_b128(gb + 8, &Bs[0][sr][so + 8]);
  }
  wait_async0();
  __syncthreads();

  int cur = 0;
  for (int kc = 0; kc < DM; kc += 32, cur ^= 1) {
    // issue async stage of next tile into the other buffer (overlaps wmma below)
    if (kc + 32 < DM) {
      const _Float16* ga = A  + (size_t)(m0 + sr) * DM + kc + 32 + so;
      const _Float16* gb = Bt + (size_t)(n0 + sr) * DM + kc + 32 + so;
      async_ld_b128(ga,     &As[cur ^ 1][sr][so]);
      async_ld_b128(ga + 8, &As[cur ^ 1][sr][so + 8]);
      async_ld_b128(gb,     &Bs[cur ^ 1][sr][so]);
      async_ld_b128(gb + 8, &Bs[cur ^ 1][sr][so + 8]);
    }

    v16h af[2], bf[4];
    af[0] = frag_a(&As[cur][wm * 32 + l15][0], 0, hh);
    af[1] = frag_a(&As[cur][wm * 32 + 16 + l15][0], 0, hh);
#pragma unroll
    for (int ni = 0; ni < 4; ni++)
      bf[ni] = frag_b(&Bs[cur][wn * 64 + ni * 16 + l15][0], 0, hh);
#pragma unroll
    for (int mi = 0; mi < 2; mi++)
#pragma unroll
      for (int ni = 0; ni < 4; ni++)
        acc[mi][ni] = wmma_f16(af[mi], bf[ni], acc[mi][ni]);

    wait_async0();
    __syncthreads();
  }

  // epilogue: C layout -> row = r + hh*8, col = l15 (per 16x16 subtile)
#pragma unroll
  for (int mi = 0; mi < 2; mi++)
#pragma unroll
    for (int ni = 0; ni < 4; ni++) {
      int rowb = m0 + wm * 32 + mi * 16 + hh * 8;
      int col  = n0 + wn * 64 + ni * 16 + l15;
      float bc = bias[col];
#pragma unroll
      for (int r = 0; r < 8; r++) {
        int row  = rowb + r;
        float vv = acc[mi][ni][r] + bc;
        if (MODE == 0) {
          vv *= scale;
          int b = row >> 11, s = row & (SS - 1);
          int h = col >> 6,  d = col & (DK - 1);
          ((_Float16*)outp)[(((size_t)(b * NH + h)) * SS + s) * DK + d] = (_Float16)vv;
        } else {
          ((float*)outp)[(size_t)row * DM + col] = vv;
        }
      }
    }
}

// ---------------- flash attention kernel --------------------------------------
// grid: (SS/64, BB*NH), block: 128 (4 waves, 16 q-rows each). Online softmax.
// DOUBLE-BUFFERED K/V staging: async K DMA + V global loads for tile i+1 are
// issued before the tile-i WMMA/softmax work; V transpose stores happen after.
__global__ __launch_bounds__(128) void attn_kernel(const _Float16* __restrict__ Qh,
                                                   const _Float16* __restrict__ Kh,
                                                   const _Float16* __restrict__ Vh,
                                                   _Float16* __restrict__ attn16) {
  __shared__ _Float16 Ks[2][64][72];     // K tile row-major (k-contiguous per key row)
  __shared__ _Float16 Vt[2][64][72];     // V tile transposed: Vt[d][kv_row]
  __shared__ _Float16 Pt[4][16][72];     // per-wave P tile (A-layout staging)

  const int bh   = blockIdx.y;           // b*NH + h
  const int q0   = blockIdx.x * 64;
  const int tid  = threadIdx.x;
  const int lane = tid & 31;
  const int wave = tid >> 5;             // 0..3
  const int hh   = lane >> 4;
  const int l15  = lane & 15;
  const size_t base = (size_t)bh * SS * DK;

  const int sr = tid >> 1;               // staging row 0..63
  const int so = (tid & 1) * 32;         // staging col offset (halfs)

  // Q fragments for this wave's 16 rows (Q already scaled by 1/sqrt(DK))
  const _Float16* qrow = Qh + base + (size_t)(q0 + wave * 16 + l15) * DK;
  const v16h aq0 = frag_a(qrow, 0, hh);
  const v16h aq1 = frag_a(qrow, 32, hh);

  v8f oacc[4] = {};
  float mrow[8], lrow[8];
#pragma unroll
  for (int r = 0; r < 8; r++) { mrow[r] = -1e30f; lrow[r] = 0.0f; }

  // prologue: stage tile 0 into buffer 0
  {
    const _Float16* gk = Kh + base + (size_t)sr * DK + so;
    async_ld_b128(gk,      &Ks[0][sr][so]);
    async_ld_b128(gk + 8,  &Ks[0][sr][so + 8]);
    async_ld_b128(gk + 16, &Ks[0][sr][so + 16]);
    async_ld_b128(gk + 24, &Ks[0][sr][so + 24]);
    const v8h* sv = (const v8h*)(Vh + base + (size_t)sr * DK + so);
    v8h r0 = sv[0], r1 = sv[1], r2 = sv[2], r3 = sv[3];
#pragma unroll
    for (int j = 0; j < 8; j++) {
      Vt[0][so + j][sr]      = r0[j];
      Vt[0][so + 8 + j][sr]  = r1[j];
      Vt[0][so + 16 + j][sr] = r2[j];
      Vt[0][so + 24 + j][sr] = r3[j];
    }
  }

  int cur = 0;
  for (int kv = 0; kv < SS; kv += 64, cur ^= 1) {
    wait_async0();
    __syncthreads();   // tile `cur` ready; all readers of cur^1 from iter-1 done

    const bool havenext = (kv + 64) < SS;
    v8h vr0, vr1, vr2, vr3;
    if (havenext) {
      // issue async K stage + V global loads for tile i+1 (no wait yet)
      const _Float16* gk = Kh + base + (size_t)(kv + 64 + sr) * DK + so;
      async_ld_b128(gk,      &Ks[cur ^ 1][sr][so]);
      async_ld_b128(gk + 8,  &Ks[cur ^ 1][sr][so + 8]);
      async_ld_b128(gk + 16, &Ks[cur ^ 1][sr][so + 16]);
      async_ld_b128(gk + 24, &Ks[cur ^ 1][sr][so + 24]);
      const v8h* sv = (const v8h*)(Vh + base + (size_t)(kv + 64 + sr) * DK + so);
      vr0 = sv[0]; vr1 = sv[1]; vr2 = sv[2]; vr3 = sv[3];
    }

    // scores: K fragments from LDS (short latency), 8 back-to-back wmma
    v16h kf[4][2];
#pragma unroll
    for (int nt = 0; nt < 4; nt++) {
      const _Float16* krow = &Ks[cur][nt * 16 + l15][0];
      kf[nt][0] = frag_b(krow, 0, hh);
      kf[nt][1] = frag_b(krow, 32, hh);
    }
    v8f sacc[4] = {};
#pragma unroll
    for (int nt = 0; nt < 4; nt++) {
      sacc[nt] = wmma_f16(aq0, kf[nt][0], sacc[nt]);
      sacc[nt] = wmma_f16(aq1, kf[nt][1], sacc[nt]);
    }

    // online softmax (rows live on 16-lane halves; DPP row-rotation reductions)
#pragma unroll
    for (int r = 0; r < 8; r++) {
      float pm = fmaxf(fmaxf(sacc[0][r], sacc[1][r]), fmaxf(sacc[2][r], sacc[3][r]));
      pm = rowred_max(pm);
      float mnew  = fmaxf(mrow[r], pm);
      float alpha = __expf(mrow[r] - mnew);
      float rsum  = 0.0f;
#pragma unroll
      for (int nt = 0; nt < 4; nt++) {
        float p = __expf(sacc[nt][r] - mnew);
        sacc[nt][r] = p;
        rsum += p;
      }
      rsum = rowred_sum(rsum);
      lrow[r] = lrow[r] * alpha + rsum;
      mrow[r] = mnew;
#pragma unroll
      for (int dt = 0; dt < 4; dt++) oacc[dt][r] *= alpha;
    }

    // C-layout -> A-layout for P via per-wave LDS bounce (f16)
#pragma unroll
    for (int nt = 0; nt < 4; nt++)
#pragma unroll
      for (int r = 0; r < 8; r++)
        Pt[wave][r + hh * 8][nt * 16 + l15] = (_Float16)sacc[nt][r];
    asm volatile("s_wait_dscnt 0x0" ::: "memory");

    // O += P @ V : preload P A-frags and V B-frags from LDS, 8 back-to-back wmma
    const _Float16* prow = &Pt[wave][l15][0];
    const v16h ap0 = frag_a(prow, 0, hh);
    const v16h ap1 = frag_a(prow, 32, hh);
    v16h vf[4][2];
#pragma unroll
    for (int dt = 0; dt < 4; dt++) {
      const _Float16* vrow = &Vt[cur][dt * 16 + l15][0];
      vf[dt][0] = frag_b(vrow, 0, hh);
      vf[dt][1] = frag_b(vrow, 32, hh);
    }
#pragma unroll
    for (int dt = 0; dt < 4; dt++) {
      oacc[dt] = wmma_f16(ap0, vf[dt][0], oacc[dt]);
      oacc[dt] = wmma_f16(ap1, vf[dt][1], oacc[dt]);
    }

    // transpose-store the prefetched V registers for tile i+1 (latency hidden
    // behind the wmma/softmax work above)
    if (havenext) {
#pragma unroll
      for (int j = 0; j < 8; j++) {
        Vt[cur ^ 1][so + j][sr]      = vr0[j];
        Vt[cur ^ 1][so + 8 + j][sr]  = vr1[j];
        Vt[cur ^ 1][so + 16 + j][sr] = vr2[j];
        Vt[cur ^ 1][so + 24 + j][sr] = vr3[j];
      }
    }
  }

  // finalize: attn16 row-major [B*S, NH*DK] (column block = h*64)
  const int b = bh >> 4, h = bh & (NH - 1);
#pragma unroll
  for (int dt = 0; dt < 4; dt++) {
#pragma unroll
    for (int r = 0; r < 8; r++) {
      int row = q0 + wave * 16 + r + hh * 8;
      int c   = dt * 16 + l15;
      float v = oacc[dt][r] * (1.0f / lrow[r]);
      attn16[(size_t)(b * SS + row) * DM + h * DK + c] = (_Float16)v;
    }
  }
}

// ---------------- launch ------------------------------------------------------

extern "C" void kernel_launch(void* const* d_in, const int* in_sizes, int n_in,
                              void* d_out, int out_size, void* d_ws, size_t ws_size,
                              hipStream_t stream) {
  const float* q  = (const float*)d_in[0];
  const float* k  = (const float*)d_in[1];
  const float* v  = (const float*)d_in[2];
  const float* Wq = (const float*)d_in[3];
  const float* bq = (const float*)d_in[4];
  const float* Wk = (const float*)d_in[5];
  const float* bk = (const float*)d_in[6];
  const float* Wv = (const float*)d_in[7];
  const float* bv = (const float*)d_in[8];
  const float* Wo = (const float*)d_in[9];
  const float* bo = (const float*)d_in[10];

  const size_t ACT = (size_t)MROWS * DM;   // 8.39M halfs
  const size_t WSZ = (size_t)DM * DM;      // 1.05M halfs

  _Float16* w   = (_Float16*)d_ws;
  _Float16* q16 = w;
  _Float16* k16 = q16 + ACT;
  _Float16* v16 = k16 + ACT;
  _Float16* wqT = v16 + ACT;
  _Float16* wkT = wqT + WSZ;
  _Float16* wvT = wkT + WSZ;
  _Float16* woT = wvT + WSZ;
  _Float16* Qh  = woT + WSZ;
  _Float16* Kh  = Qh + ACT;
  _Float16* Vh  = Kh + ACT;
  _Float16* a16 = q16;                     // reuse q16 after Q projection is done

  const int cvtBlocks = (int)(ACT / 4 / 256);          // 8192
  cvt_f16_kernel<<<cvtBlocks, 256, 0, stream>>>(q, q16, (int)ACT);
  cvt_f16_kernel<<<cvtBlocks, 256, 0, stream>>>(k, k16, (int)ACT);
  cvt_f16_kernel<<<cvtBlocks, 256, 0, stream>>>(v, v16, (int)ACT);

  const int tBlocks = (int)(WSZ / 256);                // 4096
  cvtT_f16_kernel<<<tBlocks, 256, 0, stream>>>(Wq, wqT);
  cvtT_f16_kernel<<<tBlocks, 256, 0, stream>>>(Wk, wkT);
  cvtT_f16_kernel<<<tBlocks, 256, 0, stream>>>(Wv, wvT);
  cvtT_f16_kernel<<<tBlocks, 256, 0, stream>>>(Wo, woT);

  dim3 gg(MROWS / 128, DM / 128);                      // (64, 8)
  gemm_kernel<0><<<gg, 256, 0, stream>>>(q16, wqT, bq, (void*)Qh, 0.125f);
  gemm_kernel<0><<<gg, 256, 0, stream>>>(k16, wkT, bk, (void*)Kh, 1.0f);
  gemm_kernel<0><<<gg, 256, 0, stream>>>(v16, wvT, bv, (void*)Vh, 1.0f);

  dim3 ga(SS / 64, BB * NH);                           // (32, 64)
  attn_kernel<<<ga, 128, 0, stream>>>(Qh, Kh, Vh, a16);

  gemm_kernel<1><<<gg, 256, 0, stream>>>(a16, woT, bo, d_out, 1.0f);
}
